// MultiHeadAttentionLayer_33354716020984
// MI455X (gfx1250) — matrix operand order repaired and checked by reading, not compiled, not run on hardware
//
#include <hip/hip_runtime.h>

// ---------------- Problem constants ----------------
#define B_    4
#define S_    2048
#define DIN   1024
#define NH    16
#define DK    64

typedef __attribute__((ext_vector_type(16))) __bf16 v16bf;
typedef __attribute__((ext_vector_type(8)))  __bf16 v8bf;
typedef __attribute__((ext_vector_type(8)))  float  v8f;
typedef __attribute__((ext_vector_type(4)))  unsigned int u32x4;
typedef __attribute__((ext_vector_type(8)))  int i32x8;
typedef __attribute__((ext_vector_type(4)))  int i32x4;

// ---------------- WMMA fragment loaders (wave32, 16x16x32 bf16) ----------------
// A (16x32, MxK), row-major source, leading dim `ld` elements.
// Lane L: row = L&15 ; elem i -> K=(L>>4)*16+i
__device__ __forceinline__ v16bf load_A_frag(const __bf16* base, int ld) {
  const int lane = threadIdx.x & 31;
  const __bf16* p = base + (size_t)(lane & 15) * ld + ((lane >> 4) << 4);
  return *(const v16bf*)(p);
}

// B (32x16, KxN) from K-major storage BT[n][k], leading dim `ld` elements.
// Lane L: col = L&15 ; elem i -> K=(L>>4)*16+i (contiguous 32B per lane)
__device__ __forceinline__ v16bf load_B_frag(const __bf16* base, long long ld) {
  const int lane = threadIdx.x & 31;
  const __bf16* p = base + (long long)(lane & 15) * ld + ((lane >> 4) << 4);
  return *(const v16bf*)(p);
}

__device__ __forceinline__ v8f wmma_bf16(v16bf a, v16bf b, v8f c) {
  return __builtin_amdgcn_wmma_f32_16x16x32_bf16(false, a, false, b, (short)0, c,
                                                 false, false);
}

// ---------------- Tensor Data Mover: 2-D tile Global -> LDS ----------------
// D# per CDNA5 ISA sect.8. 2-byte elements, 2-D tile,
// extra descriptor groups zero (VADDR2/3 disabled-equivalent), TENSORcnt-tracked.
// This toolchain exposes the 6-arg builtin:
//   (uint32x4 g0, int32x8 g1, int32x4, int32x4, int32x8, i32 cpol)
__device__ __forceinline__ void tdm_load_2d(void* lds_dst, const void* gsrc,
                                            unsigned tensor_d0, unsigned tensor_d1,
                                            unsigned tile_d0, unsigned tile_d1,
                                            unsigned long long stride0) {
  unsigned long long ga = (unsigned long long)(uintptr_t)gsrc;
  unsigned lds = (unsigned)(uintptr_t)lds_dst;  // low 32 bits of generic = LDS offset
  u32x4 g0;
  g0[0] = 1u;                                    // count=1 (valid user descriptor)
  g0[1] = lds;                                   // lds_addr
  g0[2] = (unsigned)ga;                          // global_addr[31:0]
  g0[3] = (unsigned)(ga >> 32) | 0x80000000u;    // global_addr[56:32] | type=2
  i32x8 g1;
  g1[0] = (int)(1u << 16);                       // workgroup_mask=0, data_size=1 (2B)
  g1[1] = (int)((tensor_d0 & 0xFFFFu) << 16);    // tensor_dim0[15:0]
  g1[2] = (int)((tensor_d0 >> 16) | ((tensor_d1 & 0xFFFFu) << 16));
  g1[3] = (int)((tensor_d1 >> 16) | (tile_d0 << 16));  // tile_dim0
  g1[4] = (int)tile_d1;                          // tile_dim1, tile_dim2=0
  g1[5] = (int)(unsigned)(stride0 & 0xFFFFFFFFu);       // tensor_dim0_stride[31:0]
  g1[6] = (int)(unsigned)((stride0 >> 32) & 0xFFFFu);   // stride0[47:32], stride1=0
  g1[7] = 0;
  i32x4 z4 = {0, 0, 0, 0};
  i32x8 z8 = {0, 0, 0, 0, 0, 0, 0, 0};
  __builtin_amdgcn_tensor_load_to_lds(g0, g1, z4, z4, z8, 0);
}

// ---------------- Conversion kernels ----------------
__global__ void cvt_f32_bf16_kernel(const float* __restrict__ in,
                                    __bf16* __restrict__ out, long long n) {
  long long i = (long long)blockIdx.x * blockDim.x + threadIdx.x;
  long long stride = (long long)gridDim.x * blockDim.x;
  for (; i < n; i += stride) out[i] = (__bf16)in[i];
}

// in [H][K][N] f32 -> out [H][N][K] bf16 (K-major for B-fragments)
__global__ void cvt_transpose_kernel(const float* __restrict__ in,
                                     __bf16* __restrict__ out,
                                     int H, int K, int N) {
  long long total = (long long)H * K * N;
  long long i = (long long)blockIdx.x * blockDim.x + threadIdx.x;
  long long stride = (long long)gridDim.x * blockDim.x;
  for (; i < total; i += stride) {
    int h = (int)(i / ((long long)K * N));
    int rem = (int)(i % ((long long)K * N));
    int k = rem / N;
    int n = rem % N;
    out[(long long)h * N * K + (long long)n * K + k] = (__bf16)in[i];
  }
}

// ---------------- QKV projection: out[bh] = X[b] (SxDIN) * W[h]^T + bias ----------------
// grid: (S/128, B*NH), block 128 (4 waves). Wave w -> 32-row strip, N=64.
// Weight panel [64][1024] streamed through LDS in K-slices of 128 via TDM,
// double-buffered (2 x 16 KB).
template <bool TRANS_OUT>
__global__ __launch_bounds__(128) void proj_kernel(
    const __bf16* __restrict__ X,   // [B][S][DIN] row-major
    const __bf16* __restrict__ WT,  // [NH][DK][DIN] (K-major)
    const float* __restrict__ bias, // [NH][DK]
    __bf16* __restrict__ out) {     // [BH][S][DK] or [BH][DK][S]
  __shared__ __attribute__((aligned(32))) __bf16 Bs[2][DK * 128];

  const int bh = blockIdx.y;
  const int b = bh >> 4, h = bh & 15;
  const int wave = threadIdx.x >> 5;
  const int m0 = blockIdx.x * 128 + wave * 32;

  const __bf16* Xb = X + (long long)b * S_ * DIN + (long long)m0 * DIN;
  const __bf16* Wb = WT + (long long)h * DK * DIN;  // [64][1024] K-major

  v8f acc[2][4] = {};

  // prologue: slice 0 -> Bs[0]
  if (wave == 0)
    tdm_load_2d(&Bs[0][0], Wb, DIN, DK, 128, DK, DIN);

  const int NSLICE = DIN / 128;  // 8
  for (int s = 0; s < NSLICE; ++s) {
    if (wave == 0) __builtin_amdgcn_s_wait_tensorcnt(0);
    __syncthreads();  // slice s visible to all waves
    if (wave == 0 && s + 1 < NSLICE)
      tdm_load_2d(&Bs[(s + 1) & 1][0], Wb + (s + 1) * 128, DIN, DK, 128, DK, DIN);

    const __bf16* Bp = &Bs[s & 1][0];
#pragma unroll
    for (int kk = 0; kk < 128; kk += 32) {
      const int k0 = s * 128 + kk;
      __builtin_prefetch(Xb + k0 + 512, 0, 1);
      v16bf a0 = load_A_frag(Xb + k0, DIN);
      v16bf a1 = load_A_frag(Xb + 16 * DIN + k0, DIN);
#pragma unroll
      for (int t = 0; t < 4; ++t) {
        v16bf bm = load_B_frag(Bp + t * 16 * 128 + kk, 128);
        acc[0][t] = wmma_bf16(a0, bm, acc[0][t]);
        acc[1][t] = wmma_bf16(a1, bm, acc[1][t]);
      }
    }
    __syncthreads();  // all reads of Bs[s&1] done before it is overwritten
  }

  const int lane = threadIdx.x & 31;
  const int col = lane & 15;
  const int r0 = (lane >> 4) * 8;
  const float* bb = bias + h * DK;
#pragma unroll
  for (int ms = 0; ms < 2; ++ms) {
#pragma unroll
    for (int t = 0; t < 4; ++t) {
      const float bvv = bb[t * 16 + col];
#pragma unroll
      for (int r = 0; r < 8; ++r) {
        const int row = m0 + ms * 16 + r0 + r;
        float v = acc[ms][t][r] + bvv;
        if (!TRANS_OUT) {
          out[(long long)bh * S_ * DK + (long long)row * DK + t * 16 + col] = (__bf16)v;
        } else {  // V stored transposed [DK][S] for contiguous P*V B-fragments
          out[(long long)bh * DK * S_ + (long long)(t * 16 + col) * S_ + row] = (__bf16)v;
        }
      }
    }
  }
}

// ---------------- Flash attention per (b,h): O = softmax(mask(QK^T/8)) V ----------------
// grid: (S/16/4, B*NH), block 128 (4 waves, each an independent 16-row strip).
__global__ __launch_bounds__(128) void attn_kernel(
    const __bf16* __restrict__ Q,   // [BH][S][DK]
    const __bf16* __restrict__ K,   // [BH][S][DK] (row-major == K-major for QK^T)
    const __bf16* __restrict__ Vt,  // [BH][DK][S] (K-major for P*V)
    __bf16* __restrict__ A) {       // [B][S][NH*DK] head-concat
  __shared__ __attribute__((aligned(32))) __bf16 Plds[4][16 * 32];

  const int bh = blockIdx.y;
  const int b = bh >> 4, h = bh & 15;
  const int wave = threadIdx.x >> 5;
  const int itile = blockIdx.x * 4 + wave;
  const int m0 = itile * 16;

  const __bf16* Qb = Q + (long long)bh * S_ * DK + (long long)m0 * DK;
  const __bf16* Kb = K + (long long)bh * S_ * DK;
  const __bf16* Vb = Vt + (long long)bh * DK * S_;

  const int lane = threadIdx.x & 31;
  const int col = lane & 15;
  const int r0 = (lane >> 4) * 8;

  v16bf aq0 = load_A_frag(Qb, DK);
  v16bf aq1 = load_A_frag(Qb + 32, DK);

  v8f acc[4] = {};
  float mrun[8], lrun[8];
#pragma unroll
  for (int r = 0; r < 8; ++r) { mrun[r] = -1e30f; lrun[r] = 0.0f; }

  __bf16* Pw = &Plds[wave][0];
  const int nsteps = (m0 + 16 + 31) >> 5;  // causal: cols 0 .. m0+15

  for (int st = 0; st < nsteps; ++st) {
    const int c0 = st * 32;
    __builtin_prefetch(Kb + (long long)(c0 + 32) * DK, 0, 1);
    // ---- scores: two 16x16 tiles (cols c0..+15 and c0+16..+31) ----
    v8f sa = {}, sb = {};
    {
      v16bf bk0 = load_B_frag(Kb + (long long)c0 * DK, DK);
      v16bf bk1 = load_B_frag(Kb + (long long)c0 * DK + 32, DK);
      sa = wmma_bf16(aq0, bk0, sa);
      sa = wmma_bf16(aq1, bk1, sa);
      v16bf bk2 = load_B_frag(Kb + (long long)(c0 + 16) * DK, DK);
      v16bf bk3 = load_B_frag(Kb + (long long)(c0 + 16) * DK + 32, DK);
      sb = wmma_bf16(aq0, bk2, sb);
      sb = wmma_bf16(aq1, bk3, sb);
    }
    // ---- online softmax over these 32 columns ----
#pragma unroll
    for (int r = 0; r < 8; ++r) {
      const int grow = m0 + r0 + r;
      float va = sa[r] * 0.125f;
      float vb = sb[r] * 0.125f;
      if (c0 + col > grow) va = -1e30f;
      if (c0 + 16 + col > grow) vb = -1e30f;
      float rm = fmaxf(va, vb);
      rm = fmaxf(rm, __shfl_xor(rm, 1, 16));
      rm = fmaxf(rm, __shfl_xor(rm, 2, 16));
      rm = fmaxf(rm, __shfl_xor(rm, 4, 16));
      rm = fmaxf(rm, __shfl_xor(rm, 8, 16));
      const float mn = fmaxf(mrun[r], rm);
      const float pa = __expf(va - mn);
      const float pb = __expf(vb - mn);
      const float alpha = __expf(mrun[r] - mn);
      float rs = pa + pb;
      rs += __shfl_xor(rs, 1, 16);
      rs += __shfl_xor(rs, 2, 16);
      rs += __shfl_xor(rs, 4, 16);
      rs += __shfl_xor(rs, 8, 16);
      lrun[r] = lrun[r] * alpha + rs;
      mrun[r] = mn;
      Pw[(r0 + r) * 32 + col] = (__bf16)pa;
      Pw[(r0 + r) * 32 + 16 + col] = (__bf16)pb;
#pragma unroll
      for (int t = 0; t < 4; ++t) acc[t][r] *= alpha;
    }
    // ---- O += P (16x32) * V (32x64) ----
    v16bf ap = load_A_frag(Pw, 32);
#pragma unroll
    for (int t = 0; t < 4; ++t) {
      v16bf bv = load_B_frag(Vb + (long long)(t * 16) * S_ + c0, S_);
      acc[t] = wmma_bf16(ap, bv, acc[t]);
    }
  }

  // ---- normalize + store to head-concat buffer [B][S][NH*DK] (bf16) ----
  __bf16* Ob = A + (long long)b * S_ * (NH * DK) + (long long)m0 * (NH * DK) + h * DK;
#pragma unroll
  for (int r = 0; r < 8; ++r) {
    const float inv = 1.0f / lrun[r];
#pragma unroll
    for (int t = 0; t < 4; ++t)
      Ob[(long long)(r0 + r) * (NH * DK) + t * 16 + col] = (__bf16)(acc[t][r] * inv);
  }
}

// ---------------- Output projection: out = A (M x 1024) * Wo + bo (f32) ----------------
// grid: (M/128, 1024/64), block 128 (4 waves, 32-row strips).
// WoT panel [n0..n0+64) x K streamed through LDS via TDM, double-buffered.
__global__ __launch_bounds__(128) void outproj_kernel(
    const __bf16* __restrict__ A,    // [M][DIN] bf16
    const __bf16* __restrict__ WoT,  // [DIN][DIN] K-major: WoT[n][k] = Wo[k][n]
    const float* __restrict__ bo,    // [DIN]
    float* __restrict__ out) {       // [M][DIN] f32
  __shared__ __attribute__((aligned(32))) __bf16 Bs[2][64 * 128];

  const int wave = threadIdx.x >> 5;
  const int m0 = blockIdx.x * 128 + wave * 32;
  const int n0 = blockIdx.y * 64;

  const __bf16* Ab = A + (long long)m0 * DIN;
  const __bf16* Wb = WoT + (long long)n0 * DIN;  // [64][1024] K-major

  v8f acc[2][4] = {};

  if (wave == 0)
    tdm_load_2d(&Bs[0][0], Wb, DIN, 64, 128, 64, DIN);

  const int NSLICE = DIN / 128;  // 8
  for (int s = 0; s < NSLICE; ++s) {
    if (wave == 0) __builtin_amdgcn_s_wait_tensorcnt(0);
    __syncthreads();
    if (wave == 0 && s + 1 < NSLICE)
      tdm_load_2d(&Bs[(s + 1) & 1][0], Wb + (s + 1) * 128, DIN, 64, 128, 64, DIN);

    const __bf16* Bp = &Bs[s & 1][0];
#pragma unroll
    for (int kk = 0; kk < 128; kk += 32) {
      const int k0 = s * 128 + kk;
      __builtin_prefetch(Ab + k0 + 512, 0, 1);
      v16bf a0 = load_A_frag(Ab + k0, DIN);
      v16bf a1 = load_A_frag(Ab + 16 * DIN + k0, DIN);
#pragma unroll
      for (int t = 0; t < 4; ++t) {
        v16bf bm = load_B_frag(Bp + t * 16 * 128 + kk, 128);
        acc[0][t] = wmma_bf16(a0, bm, acc[0][t]);
        acc[1][t] = wmma_bf16(a1, bm, acc[1][t]);
      }
    }
    __syncthreads();
  }

  const int lane = threadIdx.x & 31;
  const int col = lane & 15;
  const int r0 = (lane >> 4) * 8;
#pragma unroll
  for (int ms = 0; ms < 2; ++ms) {
#pragma unroll
    for (int t = 0; t < 4; ++t) {
      const float bias = bo[n0 + t * 16 + col];
#pragma unroll
      for (int r = 0; r < 8; ++r) {
        const int row = m0 + ms * 16 + r0 + r;
        out[(long long)row * DIN + n0 + t * 16 + col] = acc[ms][t][r] + bias;
      }
    }
  }
}

// ---------------- Host launch ----------------
extern "C" void kernel_launch(void* const* d_in, const int* in_sizes, int n_in,
                              void* d_out, int out_size, void* d_ws, size_t ws_size,
                              hipStream_t stream) {
  const float* X  = (const float*)d_in[0];
  const float* Wq = (const float*)d_in[1];
  const float* bq = (const float*)d_in[2];
  const float* Wk = (const float*)d_in[3];
  const float* bk = (const float*)d_in[4];
  const float* Wv = (const float*)d_in[5];
  const float* bv = (const float*)d_in[6];
  const float* Wo = (const float*)d_in[7];
  const float* bo = (const float*)d_in[8];
  float* out = (float*)d_out;

  char* ws = (char*)d_ws;
  size_t off = 0;
  auto take = [&](size_t elems) -> __bf16* {
    __bf16* p = (__bf16*)(ws + off);
    off += ((elems * sizeof(__bf16)) + 255) & ~(size_t)255;
    return p;
  };
  __bf16* Xbf = take((size_t)B_ * S_ * DIN);        // 16 MB
  __bf16* WqT = take((size_t)NH * DK * DIN);        // 2 MB
  __bf16* WkT = take((size_t)NH * DK * DIN);
  __bf16* WvT = take((size_t)NH * DK * DIN);
  __bf16* WoT = take((size_t)DIN * DIN);            // 2 MB
  __bf16* Qb  = take((size_t)B_ * NH * S_ * DK);    // 16 MB
  __bf16* Kb  = take((size_t)B_ * NH * S_ * DK);
  __bf16* Vtb = take((size_t)B_ * NH * S_ * DK);
  __bf16* Abf = take((size_t)B_ * S_ * (NH * DK));  // 16 MB

  // --- conversions ---
  cvt_f32_bf16_kernel<<<2048, 256, 0, stream>>>(X, Xbf, (long long)B_ * S_ * DIN);
  cvt_transpose_kernel<<<1024, 256, 0, stream>>>(Wq, WqT, NH, DIN, DK);
  cvt_transpose_kernel<<<1024, 256, 0, stream>>>(Wk, WkT, NH, DIN, DK);
  cvt_transpose_kernel<<<1024, 256, 0, stream>>>(Wv, WvT, NH, DIN, DK);
  cvt_transpose_kernel<<<1024, 256, 0, stream>>>(Wo, WoT, 1, DIN, DIN);

  // --- QKV projections (TDM-staged weight panels) ---
  dim3 pg(S_ / 128, B_ * NH);
  proj_kernel<false><<<pg, 128, 0, stream>>>(Xbf, WqT, bq, Qb);
  proj_kernel<false><<<pg, 128, 0, stream>>>(Xbf, WkT, bk, Kb);
  proj_kernel<true><<<pg, 128, 0, stream>>>(Xbf, WvT, bv, Vtb);

  // --- flash attention ---
  dim3 ag(S_ / 16 / 4, B_ * NH);
  attn_kernel<<<ag, 128, 0, stream>>>(Qb, Kb, Vtb, Abf);

  // --- output projection (TDM-staged WoT panels) ---
  dim3 og((B_ * S_) / 128, DIN / 64);
  outproj_kernel<<<og, 128, 0, stream>>>(Abf, WoT, bo, out);
}
